// RobertaSelfAttention_36309653520690
// MI455X (gfx1250) — matrix operand order repaired and checked by reading, not compiled, not run on hardware
//
#include <hip/hip_runtime.h>

typedef __bf16 bf16_t;
typedef __attribute__((ext_vector_type(16))) __bf16 v16bf;
typedef __attribute__((ext_vector_type(8)))  __bf16 v8bf;
typedef __attribute__((ext_vector_type(8)))  float  v8f;
typedef int v4i __attribute__((vector_size(16)));

#define HIDC  1024
#define SEQ   2048
#define BATCH 2
#define NHEAD 16
#define HEADD 64
#define ROWS  (BATCH*SEQ)   // 4096
#define PSTR  72            // padded LDS row stride (bf16 elems); 144B = 16B-aligned

#define GLOBAL_AS __attribute__((address_space(1)))
#define LDS_AS    __attribute__((address_space(3)))

__device__ __forceinline__ v16bf make_frag(v8bf lo, v8bf hi) {
    v16bf r;
#pragma unroll
    for (int i = 0; i < 8; ++i) { r[i] = lo[i]; r[i + 8] = hi[i]; }
    return r;
}

__device__ __forceinline__ void wait_async0() {
#if __has_builtin(__builtin_amdgcn_s_wait_asynccnt)
    __builtin_amdgcn_s_wait_asynccnt(0);
#else
    asm volatile("s_wait_asynccnt 0x0" ::: "memory");
#endif
}

// ---------------------------------------------------------------------------
// f32 -> bf16 pack
// ---------------------------------------------------------------------------
__global__ __launch_bounds__(256) void cvt_f32_bf16(const float* __restrict__ src,
                                                    bf16_t* __restrict__ dst, int n) {
    int i = (blockIdx.x * blockDim.x + threadIdx.x) * 4;
    if (i + 3 < n) {
        float4 v = *(const float4*)(src + i);
        dst[i + 0] = (bf16_t)v.x;
        dst[i + 1] = (bf16_t)v.y;
        dst[i + 2] = (bf16_t)v.z;
        dst[i + 3] = (bf16_t)v.w;
    }
}

// ---------------------------------------------------------------------------
// Fused QKV projection: Out = (X @ W^T + b) [* 0.125 for Q], bf16 out.
// Block = 256 thr (8 waves). Block tile 128(M) x 128(N); wave tile 32x64.
// ---------------------------------------------------------------------------
__global__ __launch_bounds__(256) void qkv_gemm(
    const bf16_t* __restrict__ X,
    const bf16_t* __restrict__ Wq, const bf16_t* __restrict__ Wk, const bf16_t* __restrict__ Wv,
    const float* __restrict__ bq, const float* __restrict__ bk, const float* __restrict__ bv,
    bf16_t* __restrict__ Qo, bf16_t* __restrict__ Ko, bf16_t* __restrict__ Vo) {

    const int z = blockIdx.z;
    const bf16_t* W    = (z == 0) ? Wq : ((z == 1) ? Wk : Wv);
    const float*  bias = (z == 0) ? bq : ((z == 1) ? bk : bv);
    bf16_t*       Out  = (z == 0) ? Qo : ((z == 1) ? Ko : Vo);
    const float scale = (z == 0) ? 0.125f : 1.0f;   // fold 1/sqrt(64) into Q

    const int lane = threadIdx.x & 31;
    const int wave = threadIdx.x >> 5;
    const int l16  = lane & 15;
    const int hi   = lane >> 4;            // 0: lanes 0-15, 1: lanes 16-31

    const int m0 = blockIdx.y * 128 + (wave >> 1) * 32;
    const int n0 = blockIdx.x * 128 + (wave & 1) * 64;

    v8f acc[2][4];
#pragma unroll
    for (int mi = 0; mi < 2; ++mi)
#pragma unroll
        for (int j = 0; j < 4; ++j) acc[mi][j] = (v8f)0.0f;

    for (int kc = 0; kc < HIDC / 32; ++kc) {
        // A fragments: lane holds row M=l16; K groups {base..+7, base+16..+23}
        v16bf a[2];
#pragma unroll
        for (int mi = 0; mi < 2; ++mi) {
            const bf16_t* ap = X + (size_t)(m0 + mi * 16 + l16) * HIDC + kc * 32 + hi * 8;
            a[mi] = make_frag(*(const v8bf*)ap, *(const v8bf*)(ap + 16));
        }
#pragma unroll
        for (int j = 0; j < 4; ++j) {
            // B fragment: lane holds W row (output col); K groups {base..+7, base+16..+23}
            const bf16_t* bp = W + (size_t)(n0 + j * 16 + l16) * HIDC + kc * 32 + hi * 8;
            v16bf bmat = make_frag(*(const v8bf*)bp, *(const v8bf*)(bp + 16));
#pragma unroll
            for (int mi = 0; mi < 2; ++mi)
                acc[mi][j] = __builtin_amdgcn_wmma_f32_16x16x32_bf16(
                    false, a[mi], false, bmat, (short)0, acc[mi][j], false, false);
        }
    }

    // Epilogue: bias + scale, bf16 store. C layout: VGPR r -> row r (+8 for hi half).
#pragma unroll
    for (int j = 0; j < 4; ++j) {
        const int col = n0 + j * 16 + l16;
        const float bv_ = bias[col];
#pragma unroll
        for (int mi = 0; mi < 2; ++mi)
#pragma unroll
            for (int r = 0; r < 8; ++r) {
                const int row = m0 + mi * 16 + r + hi * 8;
                Out[(size_t)row * HIDC + col] = (bf16_t)((acc[mi][j][r] + bv_) * scale);
            }
    }
}

// ---------------------------------------------------------------------------
// Flash-style attention: one block per (b, h, 64-query tile); 4 waves x 16 rows.
// K tile staged to LDS via async-to-LDS (shared by all 4 waves); V staged
// transposed; next tile prefetched while current tile computes.
// ---------------------------------------------------------------------------
__global__ __launch_bounds__(128) void attn_kernel(
    const bf16_t* __restrict__ Qm, const bf16_t* __restrict__ Km, const bf16_t* __restrict__ Vm,
    const float* __restrict__ mask, float* __restrict__ Out) {

    __shared__ bf16_t Ksh[64 * PSTR];      // K tile row-major: [key][d]
    __shared__ bf16_t VT[64 * PSTR];       // V tile transposed: [d][key]
    __shared__ bf16_t PS[4 * 16 * PSTR];   // per-wave probs: [row][key]

    const int lane = threadIdx.x & 31;
    const int wave = threadIdx.x >> 5;
    const int l16  = lane & 15;
    const int hi   = lane >> 4;

    const int b = blockIdx.z;
    const int h = blockIdx.y;
    const int q0 = blockIdx.x * 64 + wave * 16;
    const size_t rowbase = (size_t)b * SEQ;
    const int cbase = h * HEADD;

    // Q A-fragments (16 rows x 64 d), Q already pre-scaled by 0.125
    v16bf aq[2];
#pragma unroll
    for (int c = 0; c < 2; ++c) {
        const bf16_t* qp = Qm + (rowbase + q0 + l16) * HIDC + cbase + c * 32 + hi * 8;
        aq[c] = make_frag(*(const v8bf*)qp, *(const v8bf*)(qp + 16));
    }

    v8f o[4];
#pragma unroll
    for (int j = 0; j < 4; ++j) o[j] = (v8f)0.0f;
    float mrow[8], lrow[8];
#pragma unroll
    for (int r = 0; r < 8; ++r) { mrow[r] = -3.402823466e38f; lrow[r] = 0.0f; }

    for (int kt = 0; kt < SEQ / 64; ++kt) {
        __syncthreads();   // all waves done reading previous Ksh/VT
        // ---- Stage K tile (row-major) via async copy to LDS; stage V^T. ----
#pragma unroll
        for (int it = 0; it < 4; ++it) {
            int p   = threadIdx.x + it * 128;
            int key = p >> 3, ch = p & 7;
            const size_t grow = (rowbase + kt * 64 + key) * HIDC + cbase + ch * 8;

            // K: memory -> LDS, ASYNCcnt-tracked when available
#if __has_builtin(__builtin_amdgcn_global_load_async_to_lds_b128)
            __builtin_amdgcn_global_load_async_to_lds_b128(
                (GLOBAL_AS v4i*)(Km + grow),
                (LDS_AS v4i*)(Ksh + key * PSTR + ch * 8), 0, 0);
#else
            *(v8bf*)(Ksh + key * PSTR + ch * 8) = *(const v8bf*)(Km + grow);
#endif
            // V: transpose-on-store
            v8bf v = *(const v8bf*)(Vm + grow);
#pragma unroll
            for (int i = 0; i < 8; ++i) VT[(ch * 8 + i) * PSTR + key] = v[i];

            // Prefetch next key tile into cache while this one computes
            if (kt + 1 < SEQ / 64) {
                __builtin_prefetch(Km + grow + (size_t)64 * HIDC, 0, 0);
                __builtin_prefetch(Vm + grow + (size_t)64 * HIDC, 0, 0);
            }
        }
        wait_async0();
        __syncthreads();

        // ---- S = Q K^T  (16 x 64 score tile, 8 WMMAs; B-frags from LDS) ----
        v8f s[4];
#pragma unroll
        for (int j = 0; j < 4; ++j) s[j] = (v8f)0.0f;
#pragma unroll
        for (int j = 0; j < 4; ++j)
#pragma unroll
            for (int c = 0; c < 2; ++c) {
                const bf16_t* kp = Ksh + (j * 16 + l16) * PSTR + c * 32 + hi * 8;
                v16bf bk_ = make_frag(*(const v8bf*)kp, *(const v8bf*)(kp + 16));
                s[j] = __builtin_amdgcn_wmma_f32_16x16x32_bf16(
                    false, aq[c], false, bk_, (short)0, s[j], false, false);
            }

        // ---- mask + clamp + online softmax (row reductions in 16-lane halves)
        float mv[4];
#pragma unroll
        for (int j = 0; j < 4; ++j) mv[j] = mask[b * SEQ + kt * 64 + j * 16 + l16];

#pragma unroll
        for (int r = 0; r < 8; ++r) {
            float t = -3.402823466e38f;
#pragma unroll
            for (int j = 0; j < 4; ++j) {
                float x = fmaxf(s[j][r] + mv[j], -3.402823466e38f);
                s[j][r] = x;
                t = fmaxf(t, x);
            }
            t = fmaxf(t, __shfl_xor(t, 1));
            t = fmaxf(t, __shfl_xor(t, 2));
            t = fmaxf(t, __shfl_xor(t, 4));
            t = fmaxf(t, __shfl_xor(t, 8));
            const float mnew  = fmaxf(mrow[r], t);
            const float alpha = __expf(mrow[r] - mnew);
            float rs = 0.0f;
#pragma unroll
            for (int j = 0; j < 4; ++j) {
                float p = __expf(s[j][r] - mnew);
                s[j][r] = p;
                rs += p;
            }
            rs += __shfl_xor(rs, 1);
            rs += __shfl_xor(rs, 2);
            rs += __shfl_xor(rs, 4);
            rs += __shfl_xor(rs, 8);
            lrow[r] = lrow[r] * alpha + rs;
            mrow[r] = mnew;
#pragma unroll
            for (int j = 0; j < 4; ++j) o[j][r] *= alpha;
        }

        // ---- Probs: C-layout -> LDS row-major -> A-layout (per-wave region)
        bf16_t* P = PS + wave * 16 * PSTR;
#pragma unroll
        for (int j = 0; j < 4; ++j)
#pragma unroll
            for (int r = 0; r < 8; ++r)
                P[(r + hi * 8) * PSTR + j * 16 + l16] = (bf16_t)s[j][r];

        // ---- O += P V  (8 WMMAs) ----
#pragma unroll
        for (int kc = 0; kc < 2; ++kc) {
            const bf16_t* pp = P + l16 * PSTR + kc * 32 + hi * 8;
            v16bf ap = make_frag(*(const v8bf*)pp, *(const v8bf*)(pp + 16));
#pragma unroll
            for (int j2 = 0; j2 < 4; ++j2) {
                const bf16_t* vtp = VT + (j2 * 16 + l16) * PSTR + kc * 32 + hi * 8;
                v16bf bvf = make_frag(*(const v8bf*)vtp, *(const v8bf*)(vtp + 16));
                o[j2] = __builtin_amdgcn_wmma_f32_16x16x32_bf16(
                    false, ap, false, bvf, (short)0, o[j2], false, false);
            }
        }
    }

    // ---- Normalize and write [b, n, h*64+d] f32 ----
#pragma unroll
    for (int j2 = 0; j2 < 4; ++j2)
#pragma unroll
        for (int r = 0; r < 8; ++r) {
            const int qrow = q0 + r + hi * 8;
            Out[((size_t)b * SEQ + qrow) * HIDC + cbase + j2 * 16 + l16] =
                o[j2][r] / lrow[r];
        }
}

// ---------------------------------------------------------------------------
extern "C" void kernel_launch(void* const* d_in, const int* in_sizes, int n_in,
                              void* d_out, int out_size, void* d_ws, size_t ws_size,
                              hipStream_t stream) {
    const float* hs   = (const float*)d_in[0];
    const float* mask = (const float*)d_in[1];
    const float* Wq   = (const float*)d_in[2];
    const float* bq   = (const float*)d_in[3];
    const float* Wk   = (const float*)d_in[4];
    const float* bk   = (const float*)d_in[5];
    const float* Wv   = (const float*)d_in[6];
    const float* bv   = (const float*)d_in[7];
    float* out = (float*)d_out;

    char* ws = (char*)d_ws;
    bf16_t* Xb  = (bf16_t*)ws; ws += (size_t)ROWS * HIDC * 2;
    bf16_t* Wqb = (bf16_t*)ws; ws += (size_t)HIDC * HIDC * 2;
    bf16_t* Wkb = (bf16_t*)ws; ws += (size_t)HIDC * HIDC * 2;
    bf16_t* Wvb = (bf16_t*)ws; ws += (size_t)HIDC * HIDC * 2;
    bf16_t* Qb  = (bf16_t*)ws; ws += (size_t)ROWS * HIDC * 2;
    bf16_t* Kb  = (bf16_t*)ws; ws += (size_t)ROWS * HIDC * 2;
    bf16_t* Vb  = (bf16_t*)ws; ws += (size_t)ROWS * HIDC * 2;

    const int nX = ROWS * HIDC, nW = HIDC * HIDC;
    cvt_f32_bf16<<<nX / 1024, 256, 0, stream>>>(hs, Xb, nX);
    cvt_f32_bf16<<<nW / 1024, 256, 0, stream>>>(Wq, Wqb, nW);
    cvt_f32_bf16<<<nW / 1024, 256, 0, stream>>>(Wk, Wkb, nW);
    cvt_f32_bf16<<<nW / 1024, 256, 0, stream>>>(Wv, Wvb, nW);

    qkv_gemm<<<dim3(HIDC / 128, ROWS / 128, 3), 256, 0, stream>>>(
        Xb, Wqb, Wkb, Wvb, bq, bk, bv, Qb, Kb, Vb);

    attn_kernel<<<dim3(SEQ / 64, NHEAD, BATCH), 128, 0, stream>>>(
        Qb, Kb, Vb, mask, out);
}
